// SemanticMemoryBank_34007551049950
// MI455X (gfx1250) — compile-verified, matrix operand order checked by reading, not run on hardware
//
#include <hip/hip_runtime.h>
#include <hip/hip_bf16.h>
#include <math.h>

typedef __bf16 bf16;
typedef __attribute__((ext_vector_type(16))) __bf16 v16bf;
typedef __attribute__((ext_vector_type(8)))  __bf16 v8bf;
typedef __attribute__((ext_vector_type(4)))  __bf16 v4bf;
typedef __attribute__((ext_vector_type(8)))  float  v8f;

#define DDIM  768
#define MROWS 100000
#define BQ    1024
#define UPD   64
#define TOPK  32

// ---------------------------------------------------------------------------
// Tiled WMMA GEMM:  out[R x 768] = A[gather[r]] @ W + bias
// A fp32 -> bf16 while staging to LDS (float4 loads, b64 LDS stores);
// W pre-converted bf16 (b128 global loads, transposed scatter into LDS).
// Block = 128 threads (4 waves). Tile 64x64, each wave owns 16 rows x 64 cols.
// ---------------------------------------------------------------------------
template<bool OUT_BF16>
__global__ __launch_bounds__(128)
void wmma_gemm(const float* __restrict__ A, const int* __restrict__ gather,
               const bf16* __restrict__ W, const float* __restrict__ bias,
               void* __restrict__ outp, int R)
{
    __shared__ bf16 sA[64][40];   // [row][k]   (pad 40 to dodge bank conflicts)
    __shared__ bf16 sB[64][40];   // [col][k]   (W tile stored transposed)
    const int tid  = threadIdx.x;
    const int lane = tid & 31;
    const int wv   = tid >> 5;
    const int half = lane >> 4;
    const int l16  = lane & 15;
    const int rowBase = blockIdx.x * 64;
    const int colBase = blockIdx.y * 64;

    v8f acc[4];
#pragma unroll
    for (int g = 0; g < 4; g++) {
        const float b0 = bias ? bias[colBase + g * 16 + l16] : 0.0f;
#pragma unroll
        for (int i = 0; i < 8; i++) acc[g][i] = b0;   // same column per lane
    }

    for (int kb = 0; kb < DDIM; kb += 32) {
        // ---- stage A (64 rows x 32 k), float4 -> 4x bf16 per iteration ----
#pragma unroll
        for (int i = 0; i < 4; i++) {
            const int q  = i * 128 + tid;     // 512 quads
            const int r  = q >> 3;
            const int c0 = (q & 7) * 4;
            int row = rowBase + r; if (row > R - 1) row = R - 1;
            const int src = gather ? gather[row] : row;
            const float4 f = *(const float4*)&A[(size_t)src * DDIM + kb + c0];
            v4bf t; t[0] = (bf16)f.x; t[1] = (bf16)f.y; t[2] = (bf16)f.z; t[3] = (bf16)f.w;
            *(v4bf*)&sA[r][c0] = t;
        }
        // ---- stage W^T (32 k x 64 cols), 8 bf16 per global b128 load ----
#pragma unroll
        for (int i = 0; i < 2; i++) {
            const int o  = i * 128 + tid;     // 256 octets
            const int kr = o >> 3;
            const int c0 = (o & 7) * 8;
            const v8bf w = *(const v8bf*)&W[(size_t)(kb + kr) * DDIM + colBase + c0];
#pragma unroll
            for (int j = 0; j < 8; j++) sB[c0 + j][kr] = w[j];
        }
        __syncthreads();

        // A fragment: 16-bit 16x32 layout (lanes 0-15: K 0..7,16..23; 16-31: +8)
        const int ar = (wv << 4) + l16;
        v16bf af;
        {
            const v8bf x = *(const v8bf*)&sA[ar][half * 8];
            const v8bf y = *(const v8bf*)&sA[ar][16 + half * 8];
#pragma unroll
            for (int i = 0; i < 8; i++) { af[i] = x[i]; af[i + 8] = y[i]; }
        }
#pragma unroll
        for (int g = 0; g < 4; g++) {
            const int bc = g * 16 + l16;
            v16bf bfr;
            const v8bf x = *(const v8bf*)&sB[bc][half * 8];
            const v8bf y = *(const v8bf*)&sB[bc][16 + half * 8];
#pragma unroll
            for (int i = 0; i < 8; i++) { bfr[i] = x[i]; bfr[i + 8] = y[i]; }
            acc[g] = __builtin_amdgcn_wmma_f32_16x16x32_bf16(
                false, af, false, bfr, (short)0, acc[g], false, false);
        }
        __syncthreads();
    }

    // C layout: col = lane%16, row = vgpr + 8*(lane/16)
#pragma unroll
    for (int g = 0; g < 4; g++) {
        const int col = colBase + g * 16 + l16;
#pragma unroll
        for (int v = 0; v < 8; v++) {
            const int row = rowBase + (wv << 4) + v + 8 * half;
            if (row < R) {
                if (OUT_BF16) ((bf16*)outp)[(size_t)row * DDIM + col] = (bf16)acc[g][v];
                else          ((float*)outp)[(size_t)row * DDIM + col] = acc[g][v];
            }
        }
    }
}

// ---------------------------------------------------------------------------
// scores = (q @ kk^T)/sqrt(D); fused online softmax + running top-32 per row.
// Block = 128 threads owns 16 query rows; sweeps M in 64-column chunks.
// kk (bf16, 153.6 MB) is L2-resident on MI455X (192 MB L2); prefetch ahead.
// ---------------------------------------------------------------------------
__global__ __launch_bounds__(128)
void scores_softmax_topk(const bf16* __restrict__ qb, const bf16* __restrict__ kkb,
                         float* __restrict__ topw, float* __restrict__ topif,
                         int* __restrict__ topii)
{
    __shared__ bf16  sQ[16][DDIM];     // 24 KB, loaded once
    __shared__ float sS[16][64];       // score tile
    __shared__ float sTS[16][TOPK];
    __shared__ int   sTI[16][TOPK];
    const int tid  = threadIdx.x;
    const int lane = tid & 31, wv = tid >> 5;
    const int half = lane >> 4, l16 = lane & 15;
    const int row0 = blockIdx.x * 16;

    for (int e = tid; e < 16 * (DDIM / 8); e += 128) {
        const int r = e / (DDIM / 8), c = (e % (DDIM / 8)) * 8;
        *(v8bf*)&sQ[r][c] = *(const v8bf*)&qb[(size_t)(row0 + r) * DDIM + c];
    }
    for (int e = tid; e < 16 * TOPK; e += 128) { sTS[e >> 5][e & 31] = -INFINITY; sTI[e >> 5][e & 31] = 0; }
    __syncthreads();

    const float inv = rsqrtf((float)DDIM);
    float rmax = -INFINITY, rsum = 0.0f;       // online-softmax state (rows 0..15)
    float cmin = -INFINITY; int cminp = 0;     // min of current top-32 list

    const int nchunk = (MROWS + 63) / 64;
    for (int ch = 0; ch < nchunk; ch++) {
        const int mb = ch * 64;
        v8f acc;
#pragma unroll
        for (int i = 0; i < 8; i++) acc[i] = 0.0f;
        const int mcol = mb + (wv << 4) + l16;
        const size_t mrow = (size_t)((mcol < MROWS) ? mcol : (MROWS - 1)) * DDIM;

        // prefetch this lane's key row 8 chunks ahead (clamped; no divergence)
        {
            int pfc = mcol + 8 * 64; if (pfc > MROWS - 1) pfc = MROWS - 1;
            __builtin_prefetch(&kkb[(size_t)pfc * DDIM], 0, 1);
        }

        for (int kb = 0; kb < DDIM; kb += 32) {
            v16bf af, bfr;
            {
                const v8bf x = *(const v8bf*)&sQ[l16][kb + half * 8];
                const v8bf y = *(const v8bf*)&sQ[l16][kb + 16 + half * 8];
#pragma unroll
                for (int i = 0; i < 8; i++) { af[i] = x[i]; af[i + 8] = y[i]; }
            }
            {
                const v8bf x = *(const v8bf*)&kkb[mrow + kb + half * 8];
                const v8bf y = *(const v8bf*)&kkb[mrow + kb + 16 + half * 8];
#pragma unroll
                for (int i = 0; i < 8; i++) { bfr[i] = x[i]; bfr[i + 8] = y[i]; }
            }
            acc = __builtin_amdgcn_wmma_f32_16x16x32_bf16(
                false, af, false, bfr, (short)0, acc, false, false);
        }
#pragma unroll
        for (int v = 0; v < 8; v++) {
            const int rt = v + 8 * half;
            sS[rt][(wv << 4) + l16] = (mcol < MROWS) ? acc[v] * inv : -INFINITY;
        }
        __syncthreads();

        if (tid < 16) {                 // thread t scans row t of the tile
            const int t = tid;
            for (int j = 0; j < 64; j++) {
                const float s = sS[t][j];
                if (s > rmax) { rsum = rsum * __expf(rmax - s) + 1.0f; rmax = s; }
                else          { rsum += __expf(s - rmax); }
                if (s > cmin) {
                    sTS[t][cminp] = s; sTI[t][cminp] = mb + j;
                    cmin = sTS[t][0]; cminp = 0;
                    for (int q = 1; q < TOPK; q++)
                        if (sTS[t][q] < cmin) { cmin = sTS[t][q]; cminp = q; }
                }
            }
        }
        __syncthreads();
    }

    if (tid < 16) {
        const int t = tid;
        for (int a = 0; a < TOPK - 1; a++) {           // sort descending, ties by idx
            int best = a;
            for (int b = a + 1; b < TOPK; b++)
                if (sTS[t][b] > sTS[t][best] ||
                    (sTS[t][b] == sTS[t][best] && sTI[t][b] < sTI[t][best])) best = b;
            float vs = sTS[t][a]; sTS[t][a] = sTS[t][best]; sTS[t][best] = vs;
            int   vi = sTI[t][a]; sTI[t][a] = sTI[t][best]; sTI[t][best] = vi;
        }
        const size_t base = (size_t)(row0 + t) * TOPK;
        for (int j = 0; j < TOPK; j++) {
            topw[base + j]  = __expf(sTS[t][j] - rmax) / rsum;
            topif[base + j] = (float)sTI[t][j];
            topii[base + j] = sTI[t][j];
        }
    }
}

// ---------------------------------------------------------------------------
__global__ __launch_bounds__(256)
void rownorm2(const float* __restrict__ X, float* __restrict__ out)
{
    __shared__ float red[256];
    const int row = blockIdx.x;
    float s = 0.f;
    for (int c = threadIdx.x; c < DDIM; c += 256) {
        const float v = X[(size_t)row * DDIM + c]; s += v * v;
    }
    red[threadIdx.x] = s; __syncthreads();
    for (int st = 128; st > 0; st >>= 1) {
        if (threadIdx.x < st) red[threadIdx.x] += red[threadIdx.x + st];
        __syncthreads();
    }
    if (!threadIdx.x) out[row] = red[0];
}

__global__ void f32_to_bf16(const float* __restrict__ s, bf16* __restrict__ d, int n)
{
    const int i = (blockIdx.x * 256 + threadIdx.x) * 4;   // n is a multiple of 4
    if (i < n) {
        const float4 f = *(const float4*)&s[i];
        v4bf t; t[0] = (bf16)f.x; t[1] = (bf16)f.y; t[2] = (bf16)f.z; t[3] = (bf16)f.w;
        *(v4bf*)&d[i] = t;
    }
}

// ---------------------------------------------------------------------------
// Sequential update scan: 64 steps, single block. Incremental norm tracking.
// ---------------------------------------------------------------------------
__global__ __launch_bounds__(1024)
void seq_update(const float* __restrict__ nf, const float* __restrict__ imp,
                const float* __restrict__ Wg1, const float* __restrict__ bg1,
                const float* __restrict__ Wg2, const float* __restrict__ bg2,
                float* __restrict__ mk, float* __restrict__ mv,
                float* __restrict__ norm2, const float* __restrict__ fn2)
{
    __shared__ float rv[1024];
    __shared__ int   ri[1024];
    __shared__ float comb[2 * DDIM];
    __shared__ int smi, supd;
    const int tid = threadIdx.x;

    for (int step = 0; step < UPD; step++) {
        // ---- argmin over row norms (ties -> lowest index) ----
        float bvv = INFINITY; int bi = 0;
        for (int m = tid; m < MROWS; m += 1024) {
            const float v = norm2[m];
            if (v < bvv) { bvv = v; bi = m; }
        }
        rv[tid] = bvv; ri[tid] = bi; __syncthreads();
        for (int st = 512; st > 0; st >>= 1) {
            if (tid < st) {
                if (rv[tid + st] < rv[tid] ||
                    (rv[tid + st] == rv[tid] && ri[tid + st] < ri[tid])) {
                    rv[tid] = rv[tid + st]; ri[tid] = ri[tid + st];
                }
            }
            __syncthreads();
        }
        if (!tid) smi = ri[0];
        __syncthreads();
        const int mi = smi;

        // ---- comb = [f, mv[mi]] ----
        for (int e = tid; e < 2 * DDIM; e += 1024)
            comb[e] = (e < DDIM) ? nf[(size_t)step * DDIM + e]
                                 : mv[(size_t)mi * DDIM + (e - DDIM)];
        __syncthreads();

        // ---- h = relu(comb @ Wg1 + bg1); logit = h @ Wg2 + bg2 ----
        float hpart = 0.0f;
        if (tid < DDIM) {
            float h = bg1[tid];
            for (int i = 0; i < 2 * DDIM; i++) h += comb[i] * Wg1[(size_t)i * DDIM + tid];
            hpart = fmaxf(h, 0.0f) * Wg2[tid];
        }
        rv[tid] = hpart; __syncthreads();
        for (int st = 512; st > 0; st >>= 1) {
            if (tid < st) rv[tid] += rv[tid + st];
            __syncthreads();
        }
        if (!tid) supd = ((rv[0] + bg2[0]) > 0.0f) ? 1 : 0;  // sigmoid(x)>0.5 <=> x>0
        __syncthreads();

        // ---- gated scatter + incremental norm update ----
        if (supd) {
            if (tid < DDIM) {
                const float f = comb[tid];
                mk[(size_t)mi * DDIM + tid] = f;
                mv[(size_t)mi * DDIM + tid] = f * imp[step];
            }
            if (!tid) norm2[mi] = imp[step] * imp[step] * fn2[step];
        }
        __threadfence_block();
        __syncthreads();
    }
}

// ---------------------------------------------------------------------------
extern "C" void kernel_launch(void* const* d_in, const int* in_sizes, int n_in,
                              void* d_out, int out_size, void* d_ws, size_t ws_size,
                              hipStream_t stream)
{
    const float* qf   = (const float*)d_in[0];
    const float* nf   = (const float*)d_in[1];
    const float* imp  = (const float*)d_in[2];
    const float* memk = (const float*)d_in[3];
    const float* memv = (const float*)d_in[4];
    const float* Wq   = (const float*)d_in[5];
    const float* bq   = (const float*)d_in[6];
    const float* Wk   = (const float*)d_in[7];
    const float* bk   = (const float*)d_in[8];
    const float* Wv   = (const float*)d_in[9];
    const float* bv   = (const float*)d_in[10];
    const float* Wg1  = (const float*)d_in[11];
    const float* bg1  = (const float*)d_in[12];
    const float* Wg2  = (const float*)d_in[13];
    const float* bg2  = (const float*)d_in[14];

    // workspace carve-up (~158 MB): bf16 key cache + bf16 weights + norms + idx
    bf16*  kkb   = (bf16*)d_ws;
    bf16*  qb    = kkb + (size_t)MROWS * DDIM;
    bf16*  wqb   = qb  + (size_t)BQ * DDIM;
    bf16*  wkb   = wqb + (size_t)DDIM * DDIM;
    bf16*  wvb   = wkb + (size_t)DDIM * DDIM;
    float* norm2 = (float*)(wvb + (size_t)DDIM * DDIM);
    float* fn2   = norm2 + MROWS;
    int*   topii = (int*)(fn2 + UPD);

    // output layout: retrieved | top_w | top_idx | new_keys | new_values
    float* out   = (float*)d_out;
    float* retr  = out;
    float* topw  = retr + (size_t)BQ * TOPK * DDIM;
    float* topif = topw + (size_t)BQ * TOPK;
    float* nk    = topif + (size_t)BQ * TOPK;
    float* nv    = nk + (size_t)MROWS * DDIM;

    const int wn = DDIM * DDIM;
    f32_to_bf16<<<(wn / 4 + 255) / 256, 256, 0, stream>>>(Wq, wqb, wn);
    f32_to_bf16<<<(wn / 4 + 255) / 256, 256, 0, stream>>>(Wk, wkb, wn);
    f32_to_bf16<<<(wn / 4 + 255) / 256, 256, 0, stream>>>(Wv, wvb, wn);

    // q = query @ Wq + bq  (bf16 out);  kk = keys @ Wk + bk (bf16 cache)
    wmma_gemm<true><<<dim3(BQ / 64, DDIM / 64), 128, 0, stream>>>(qf, nullptr, wqb, bq, qb, BQ);
    wmma_gemm<true><<<dim3((MROWS + 63) / 64, DDIM / 64), 128, 0, stream>>>(memk, nullptr, wkb, bk, kkb, MROWS);

    // fused scores + softmax + top-k (no B x M matrix ever hits HBM)
    scores_softmax_topk<<<BQ / 16, 128, 0, stream>>>(qb, kkb, topw, topif, topii);

    // retrieved = memory_values[top_idx] @ Wv + bv  (gather-then-GEMM)
    wmma_gemm<false><<<dim3((BQ * TOPK) / 64, DDIM / 64), 128, 0, stream>>>(memv, topii, wvb, bv, retr, BQ * TOPK);

    // sequential update path
    rownorm2<<<MROWS, 256, 0, stream>>>(memv, norm2);
    rownorm2<<<UPD, 256, 0, stream>>>(nf, fn2);
    hipMemcpyAsync(nk, memk, (size_t)MROWS * DDIM * sizeof(float), hipMemcpyDeviceToDevice, stream);
    hipMemcpyAsync(nv, memv, (size_t)MROWS * DDIM * sizeof(float), hipMemcpyDeviceToDevice, stream);
    seq_update<<<1, 1024, 0, stream>>>(nf, imp, Wg1, bg1, Wg2, bg2, nk, nv, norm2, fn2);

    (void)in_sizes; (void)n_in; (void)out_size; (void)ws_size;
}
